// Router_85401129714219
// MI455X (gfx1250) — compile-verified, matrix operand order checked by reading, not run on hardware
//
#include <hip/hip_runtime.h>
#include <hip/hip_bf16.h>

// ---------------- problem constants ----------------
constexpr int B_  = 16;
constexpr int L_  = 512;
constexpr int D_  = 768;
constexpr int H_  = 12;
constexpr int K_  = 128;
constexpr int U_  = 256;
constexpr int NH_ = 4;
constexpr int FL_ = K_ + 2;        // 130 rows in final_token
constexpr int PCH_ = 8;            // q-chunks for importance pass 1
constexpr int QCHUNK_ = L_ / PCH_; // 64
constexpr int D4_ = D_ / 4;        // 192
constexpr int L4_ = L_ / 4;        // 128
constexpr int U4_ = U_ / 4;        // 64

typedef __attribute__((ext_vector_type(16))) __bf16 v16bf;
typedef __attribute__((ext_vector_type(8)))  float  v8f;

// ============================================================
// K1: att = softmax(mask over L); sentences[b,d] = sum_l att*hidden
// Weighted-sum phase vectorized: 192 threads each own one float4 of D.
// ============================================================
__global__ void sentences_kernel(const float* __restrict__ hidden,
                                 const float* __restrict__ mask,
                                 float* __restrict__ sent) {
  __shared__ float w[L_];
  __shared__ float red[256];
  const int b = blockIdx.x, tid = threadIdx.x;

  float m = -INFINITY;
  for (int l = tid; l < L_; l += 256) {
    float v = mask[b * L_ + l];
    w[l] = v;
    m = fmaxf(m, v);
  }
  red[tid] = m; __syncthreads();
  for (int s = 128; s > 0; s >>= 1) {
    if (tid < s) red[tid] = fmaxf(red[tid], red[tid + s]);
    __syncthreads();
  }
  m = red[0]; __syncthreads();

  float ssum = 0.f;
  for (int l = tid; l < L_; l += 256) {
    float e = __expf(w[l] - m);
    w[l] = e; ssum += e;
  }
  red[tid] = ssum; __syncthreads();
  for (int s = 128; s > 0; s >>= 1) {
    if (tid < s) red[tid] += red[tid + s];
    __syncthreads();
  }
  const float inv = 1.0f / red[0]; __syncthreads();
  for (int l = tid; l < L_; l += 256) w[l] *= inv;
  __syncthreads();

  if (tid < D4_) {
    const float4* hp = (const float4*)(hidden + (size_t)b * L_ * D_) + tid;
    float4 acc = make_float4(0.f, 0.f, 0.f, 0.f);
    for (int l = 0; l < L_; ++l) {
      const float4 hv = hp[(size_t)l * D4_];
      const float wl = w[l];
      acc.x += wl * hv.x; acc.y += wl * hv.y;
      acc.z += wl * hv.z; acc.w += wl * hv.w;
    }
    ((float4*)(sent + b * D_))[tid] = acc;
  }
}

// ============================================================
// K2: q[b,u] = sum_d sentences[b,d] * Wq[u,d]   (float4 row dot)
// ============================================================
__global__ void q_kernel(const float* __restrict__ sent,
                         const float* __restrict__ Wq,
                         float* __restrict__ q) {
  __shared__ __align__(16) float s[D_];
  const int b = blockIdx.x, tid = threadIdx.x;  // blockDim = 256 = U_
  for (int d = tid; d < D_; d += 256) s[d] = sent[b * D_ + d];
  __syncthreads();
  const float4* wr = (const float4*)(Wq + (size_t)tid * D_);
  const float4* s4 = (const float4*)s;
  float acc = 0.f;
  for (int d = 0; d < D4_; ++d) {
    const float4 a = s4[d], wv = wr[d];
    acc += a.x * wv.x + a.y * wv.y + a.z * wv.z + a.w * wv.w;
  }
  q[b * U_ + tid] = acc;
}

// ============================================================
// K3: K/V projections via bf16 WMMA, fp32 accumulate.
//   out[m,u] = sum_d hidden[m,d] * W[u,d],  m in [0, B*L), u in [0, U)
// ISA 16-bit A 16x32 layout: lane = M (mod 16), K-halves split at lane 16:
//   lane<16 holds K {0..7, 16..23}; lane>=16 holds K {8..15, 24..31}
// B tile (32x16) is symmetric with lane = N; since B[k,n] = W[u0+n, k],
// each lane loads a contiguous run of the W row -> b128 loads + cvt_pk.
// ============================================================
__device__ __forceinline__ v16bf load_tile16(const float* __restrict__ row,
                                             int dbase, int khalf) {
  v16bf r;
  const float* p0 = row + dbase + khalf * 8;
  const float* p1 = row + dbase + 16 + khalf * 8;
#pragma unroll
  for (int j = 0; j < 8; ++j) {
    r[j]     = (__bf16)p0[j];
    r[8 + j] = (__bf16)p1[j];
  }
  return r;
}

__global__ void __launch_bounds__(32)
kv_gemm_kernel(const float* __restrict__ hidden,
               const float* __restrict__ Wk,
               const float* __restrict__ Wv,
               float* __restrict__ kout,
               float* __restrict__ vout) {
  const int lane  = threadIdx.x;        // 0..31, one wave per block
  const int m0    = blockIdx.x * 16;    // row tile within B*L
  const int u0    = blockIdx.y * 64;    // 64 output cols per wave
  const int mr    = lane & 15;
  const int khalf = lane >> 4;

  const float* arow = hidden + (size_t)(m0 + mr) * D_;
  const float* bkrow[4];
  const float* bvrow[4];
#pragma unroll
  for (int t = 0; t < 4; ++t) {
    const int n = u0 + t * 16 + mr;
    bkrow[t] = Wk + (size_t)n * D_;
    bvrow[t] = Wv + (size_t)n * D_;
  }

  v8f ak[4] = {v8f{}, v8f{}, v8f{}, v8f{}};
  v8f av[4] = {v8f{}, v8f{}, v8f{}, v8f{}};

  for (int kb = 0; kb < D_; kb += 32) {
    const v16bf a = load_tile16(arow, kb, khalf);
#pragma unroll
    for (int t = 0; t < 4; ++t) {
      const v16bf bk = load_tile16(bkrow[t], kb, khalf);
      ak[t] = __builtin_amdgcn_wmma_f32_16x16x32_bf16(
          false, a, false, bk, (short)0, ak[t], false, false);
      const v16bf bv = load_tile16(bvrow[t], kb, khalf);
      av[t] = __builtin_amdgcn_wmma_f32_16x16x32_bf16(
          false, a, false, bv, (short)0, av[t], false, false);
    }
  }

  // C/D layout: lane holds col n = lane&15, rows m = khalf*8 + r
#pragma unroll
  for (int t = 0; t < 4; ++t) {
#pragma unroll
    for (int r = 0; r < 8; ++r) {
      const int row = m0 + khalf * 8 + r;
      const int col = u0 + t * 16 + mr;
      kout[(size_t)row * U_ + col] = ak[t][r];
      vout[(size_t)row * U_ + col] = av[t][r];
    }
  }
}

// ============================================================
// K4: per (b, head) single-query attention over L keys
// ============================================================
__global__ void attn_kernel(const float* __restrict__ qbuf,
                            const float* __restrict__ kbuf,
                            const float* __restrict__ vbuf,
                            const float* __restrict__ mask,
                            float* __restrict__ attn_out) {
  __shared__ __align__(16) float qh[64];
  __shared__ float p[L_];
  __shared__ float red[64];
  const int blk = blockIdx.x;
  const int b = blk / NH_, h = blk % NH_;
  const int tid = threadIdx.x;           // blockDim = 64
  const int off = h * 64;

  qh[tid] = qbuf[b * U_ + off + tid];
  __syncthreads();

  const float scale = 0.03608439182435161f; // 1/sqrt(768)
  const float4* q4 = (const float4*)qh;
  float lmax = -INFINITY;
  for (int l = tid; l < L_; l += 64) {
    const float4* kr = (const float4*)(kbuf + ((size_t)b * L_ + l) * U_ + off);
    float s = 0.f;
    for (int j = 0; j < 16; ++j) {
      const float4 a = q4[j], kv = kr[j];
      s += a.x * kv.x + a.y * kv.y + a.z * kv.z + a.w * kv.w;
    }
    s *= scale;
    if (mask[b * L_ + l] < -10.0f) s = -INFINITY;
    p[l] = s;
    lmax = fmaxf(lmax, s);
  }
  red[tid] = lmax; __syncthreads();
  for (int s2 = 32; s2 > 0; s2 >>= 1) {
    if (tid < s2) red[tid] = fmaxf(red[tid], red[tid + s2]);
    __syncthreads();
  }
  lmax = red[0]; __syncthreads();

  float lsum = 0.f;
  for (int l = tid; l < L_; l += 64) {
    float e = __expf(p[l] - lmax);
    p[l] = e; lsum += e;
  }
  red[tid] = lsum; __syncthreads();
  for (int s2 = 32; s2 > 0; s2 >>= 1) {
    if (tid < s2) red[tid] += red[tid + s2];
    __syncthreads();
  }
  const float inv = 1.0f / red[0]; __syncthreads();

  float acc = 0.f;
  const float* vp = vbuf + (size_t)b * L_ * U_ + off + tid;
  for (int l = 0; l < L_; ++l) acc += p[l] * vp[(size_t)l * U_];
  attn_out[b * U_ + off + tid] = acc * inv;
}

// ============================================================
// K5: new_token[b,d] = sum_u out[b,u]*Wo[d,u] + bo[d]  (float4 row dot)
// ============================================================
__global__ void newtok_kernel(const float* __restrict__ attn_out,
                              const float* __restrict__ Wo,
                              const float* __restrict__ bo,
                              float* __restrict__ ntok) {
  __shared__ __align__(16) float o[U_];
  const int b = blockIdx.x, tid = threadIdx.x;
  o[tid] = attn_out[b * U_ + tid];
  __syncthreads();
  const float4* o4 = (const float4*)o;
  for (int d = tid; d < D_; d += 256) {
    const float4* wr = (const float4*)(Wo + (size_t)d * U_);
    float acc = bo[d];
    for (int u = 0; u < U4_; ++u) {
      const float4 a = o4[u], wv = wr[u];
      acc += a.x * wv.x + a.y * wv.y + a.z * wv.z + a.w * wv.w;
    }
    ntok[b * D_ + d] = acc;
  }
}

// ============================================================
// K6: importance pass 1 — b128 streaming over the 201 MB tensor.
// 128 threads: thread t owns float4 chunk t of the 512-wide row.
// partial4[(b*H + h)*PCH + qc][t] = sum_{q in chunk} scores4[b,h,q,t]
// ============================================================
__global__ void __launch_bounds__(128)
imp_pass1(const float* __restrict__ scores, float* __restrict__ partial) {
  const int bh = blockIdx.x;        // 0..B*H-1
  const int qc = blockIdx.y;        // 0..PCH-1
  const int tid = threadIdx.x;      // 0..127
  const float4* base =
      (const float4*)(scores + ((size_t)bh * L_ + (size_t)qc * QCHUNK_) * L_) + tid;
  float4 acc = make_float4(0.f, 0.f, 0.f, 0.f);
  for (int q = 0; q < QCHUNK_; ++q) {
    const float4 v = base[(size_t)q * L4_];
    acc.x += v.x; acc.y += v.y; acc.z += v.z; acc.w += v.w;
  }
  ((float4*)(partial + ((size_t)bh * PCH_ + qc) * L_))[tid] = acc;
}

// ============================================================
// K7: finish importance, exact top-K (stable ties by index, matches
// lax.top_k), then emit indices in ascending order via prefix counts.
// ============================================================
__global__ void topk_kernel(const float* __restrict__ partial,
                            int* __restrict__ sidx) {
  __shared__ __align__(16) float imp[L_];
  __shared__ unsigned char sel[L_];
  const int b = blockIdx.x, tid = threadIdx.x;   // 256 threads

  if (tid < L4_) {
    const float4* pbase = (const float4*)(partial + (size_t)b * H_ * PCH_ * L_) + tid;
    float4 acc = make_float4(0.f, 0.f, 0.f, 0.f);
    for (int c = 0; c < H_ * PCH_; ++c) {
      const float4 v = pbase[(size_t)c * L4_];
      acc.x += v.x; acc.y += v.y; acc.z += v.z; acc.w += v.w;
    }
    const float s = 1.0f / (float)H_;
    acc.x *= s; acc.y *= s; acc.z *= s; acc.w *= s;
    ((float4*)imp)[tid] = acc;
  }
  __syncthreads();

  for (int l = tid; l < L_; l += 256) {
    const float v = imp[l];
    int rank = 0;
    for (int j = 0; j < L_; ++j) {
      const float u = imp[j];
      rank += (u > v) || (u == v && j < l);
    }
    sel[l] = (rank < K_) ? 1 : 0;
  }
  __syncthreads();

  for (int l = tid; l < L_; l += 256) {
    if (sel[l]) {
      int pos = 0;
      for (int j = 0; j < l; ++j) pos += sel[j];
      sidx[b * K_ + pos] = l;   // ascending by construction
    }
  }
}

// ============================================================
// K8: assemble final_token (B,130,D) and final_attention_mask (B,130)
// One block per output row; 192 lanes do the whole row in float4s.
// ============================================================
__global__ void assemble_kernel(const float* __restrict__ hidden,
                                const float* __restrict__ mask,
                                const float* __restrict__ ntok,
                                const int* __restrict__ sidx,
                                float* __restrict__ out_tok,
                                float* __restrict__ out_mask) {
  const int row = blockIdx.x;      // 0..B*FL-1
  const int b = row / FL_, t = row % FL_;
  const int tid = threadIdx.x;     // 256

  const float* src;
  int gidx = -1;
  if (t == 0) {
    src = hidden + (size_t)b * L_ * D_;
  } else if (t <= K_) {
    gidx = sidx[b * K_ + (t - 1)];
    src = hidden + ((size_t)b * L_ + gidx) * D_;
  } else {
    src = ntok + (size_t)b * D_;
  }
  if (tid < D4_) {
    ((float4*)(out_tok + (size_t)row * D_))[tid] = ((const float4*)src)[tid];
  }
  if (tid == 0) {
    float mv = 0.0f;
    if (gidx >= 0) mv = mask[b * L_ + gidx];
    out_mask[row] = mv;
  }
}

// ============================================================
extern "C" void kernel_launch(void* const* d_in, const int* in_sizes, int n_in,
                              void* d_out, int out_size, void* d_ws, size_t ws_size,
                              hipStream_t stream) {
  (void)in_sizes; (void)n_in; (void)out_size; (void)ws_size;

  const float* hidden = (const float*)d_in[0];   // B,L,D
  const float* amask  = (const float*)d_in[1];   // B,1,1,L
  const float* sas    = (const float*)d_in[2];   // B,H,L,L
  const float* Wq     = (const float*)d_in[3];   // U,D
  const float* Wk     = (const float*)d_in[4];   // U,D
  const float* Wv     = (const float*)d_in[5];   // U,D
  const float* Wo     = (const float*)d_in[6];   // D,U
  const float* bo     = (const float*)d_in[7];   // D

  float* out_tok  = (float*)d_out;                        // B*FL*D
  float* out_mask = out_tok + (size_t)B_ * FL_ * D_;      // B*FL

  // ---- workspace layout (bytes) ----
  char* ws = (char*)d_ws;
  size_t off = 0;
  auto alloc = [&](size_t bytes) {
    void* p = ws + off;
    off = (off + bytes + 255) & ~(size_t)255;
    return p;
  };
  float* sent     = (float*)alloc((size_t)B_ * D_ * 4);
  float* qbuf     = (float*)alloc((size_t)B_ * U_ * 4);
  float* kbuf     = (float*)alloc((size_t)B_ * L_ * U_ * 4);
  float* vbuf     = (float*)alloc((size_t)B_ * L_ * U_ * 4);
  float* attn_out = (float*)alloc((size_t)B_ * U_ * 4);
  float* ntok     = (float*)alloc((size_t)B_ * D_ * 4);
  float* partial  = (float*)alloc((size_t)B_ * H_ * PCH_ * L_ * 4);
  int*   sidx     = (int*)  alloc((size_t)B_ * K_ * 4);

  sentences_kernel<<<B_, 256, 0, stream>>>(hidden, amask, sent);
  q_kernel<<<B_, 256, 0, stream>>>(sent, Wq, qbuf);
  kv_gemm_kernel<<<dim3(B_ * L_ / 16, U_ / 64), 32, 0, stream>>>(hidden, Wk, Wv,
                                                                 kbuf, vbuf);
  attn_kernel<<<B_ * NH_, 64, 0, stream>>>(qbuf, kbuf, vbuf, amask, attn_out);
  newtok_kernel<<<B_, 256, 0, stream>>>(attn_out, Wo, bo, ntok);
  imp_pass1<<<dim3(B_ * H_, PCH_), 128, 0, stream>>>(sas, partial);
  topk_kernel<<<B_, 256, 0, stream>>>(partial, sidx);
  assemble_kernel<<<B_ * FL_, 256, 0, stream>>>(hidden, amask, ntok, sidx,
                                                out_tok, out_mask);
}
